// CausalSelfAttention_63866163691707
// MI455X (gfx1250) — matrix operand order repaired and checked by reading, not compiled, not run on hardware
//
#include <hip/hip_runtime.h>
#include <hip/hip_bf16.h>
#include <math.h>

// ---------------------------------------------------------------------------
// CDNA5 (gfx1250) causal self-attention: f16 WMMA, double-buffered LDS tiles,
// bank-conflict-free padded LDS strides, batched ds_load -> wmma issue.
// ---------------------------------------------------------------------------

typedef _Float16 h8   __attribute__((ext_vector_type(8)));
typedef _Float16 v16h __attribute__((ext_vector_type(16)));
typedef float    v8f  __attribute__((ext_vector_type(8)));

#define B_    2
#define S_    2048
#define D_    1024
#define H_    16
#define HD_   64
#define MTOT  (B_ * S_)      // 4096

#define KSTEP 64             // GEMM K-panel depth
#define KPAD  72             // padded LDS row stride (36 dwords: conflict-free)
#define VPAD  40             // padded 32-half rows (20 dwords: conflict-free)

static __device__ __forceinline__ v8f wmma_f16(v16h a, v16h b, v8f c) {
  return __builtin_amdgcn_wmma_f32_16x16x32_f16(
      false, a, false, b, (short)0, c, false, false);
}

static __device__ __forceinline__ v16h cat8(h8 lo, h8 hi) {
  return __builtin_shufflevector(lo, hi, 0,1,2,3,4,5,6,7,8,9,10,11,12,13,14,15);
}

// A fragment (16x32 f16): rows m0..m0+15 of row-major src[ld], K = kb..kb+31.
static __device__ __forceinline__ v16h load_a_frag(const _Float16* src, int ld,
                                                   int m0, int kb, int lane) {
  const int mr  = lane & 15;
  const int off = (lane >> 4) * 16;
  const _Float16* p = src + (size_t)(m0 + mr) * ld + kb;
  h8 lo = *(const h8*)(p + off);
  h8 hi = *(const h8*)(p + 8 + off);
  return cat8(lo, hi);
}

// B fragment (32x16 f16) from an LDS/row source: lane n = lane&15 supplies one
// column; 16 contiguous K starting at 16*(lane>>4).
static __device__ __forceinline__ v16h load_b_lds(const _Float16* rowbase) {
  h8 lo = *(const h8*)(rowbase);
  h8 hi = *(const h8*)(rowbase + 8);
  return cat8(lo, hi);
}

// ---------------------------------------------------------------------------
__global__ void cvt_f32_f16(const float* __restrict__ in,
                            _Float16* __restrict__ out, int n) {
  int i = blockIdx.x * 256 + threadIdx.x;
  if (i < n) out[i] = (_Float16)in[i];
}

// ---------------------------------------------------------------------------
// GEMM: Y[m][n] = sum_k A[m][k] * W[n][k] + bias[n]   (M=4096, N=K=1024)
// 4-wave workgroup computes 64x128. 128x64 B panel double-buffered in LDS:
// one barrier per 64-wide K step, global prefetch overlapped with 16 WMMAs.
// mode 0: f16 -> Q/K layout [B,H,S,64]; mode 1: f16 -> V^T [B,H,64,S];
// mode 2: f32 -> row-major [M,N]
__global__ __launch_bounds__(128)
void gemm_wg(const _Float16* __restrict__ A, const _Float16* __restrict__ W,
             const float* __restrict__ bias, _Float16* __restrict__ outh,
             float* __restrict__ outf, int mode) {
  __shared__ __align__(16) _Float16 Bs[2][128 * KPAD];   // 2 x 18 KB

  const int NT128 = D_ / 128;                            // 8
  const int wgm = blockIdx.x / NT128;
  const int wgn = blockIdx.x % NT128;
  const int tid  = threadIdx.x;
  const int wave = tid >> 5;
  const int lane = tid & 31;
  const int m0 = wgm * 64 + wave * 16;
  const int n0 = wgn * 128;
  const int NSTEP = D_ / KSTEP;                          // 16

  // stage panel 0
  {
    const _Float16* gp = W + (size_t)(n0 + tid) * D_;
    h8* lp = (h8*)&Bs[0][tid * KPAD];
#pragma unroll
    for (int i = 0; i < KSTEP / 8; ++i) lp[i] = *(const h8*)(gp + i * 8);
  }
  __syncthreads();

  v8f acc[8];
#pragma unroll
  for (int t = 0; t < 8; ++t) acc[t] = (v8f){0.f,0.f,0.f,0.f,0.f,0.f,0.f,0.f};

  for (int ks = 0; ks < NSTEP; ++ks) {
    const int cur = ks & 1;
    const int kb  = ks * KSTEP;
    const bool more = (ks + 1 < NSTEP);

    // prefetch next B panel into registers (overlaps the WMMA chain below)
    h8 pre[KSTEP / 8];
    if (more) {
      const _Float16* gp = W + (size_t)(n0 + tid) * D_ + kb + KSTEP;
#pragma unroll
      for (int i = 0; i < KSTEP / 8; ++i) pre[i] = *(const h8*)(gp + i * 8);
    }

    // compute: two 32-wide k sub-steps, B frags batched 4 at a time
#pragma unroll
    for (int kh = 0; kh < KSTEP; kh += 32) {
      const v16h a = load_a_frag(A, D_, m0, kb + kh, lane);
#pragma unroll
      for (int tb = 0; tb < 8; tb += 4) {
        v16h bf[4];
#pragma unroll
        for (int t = 0; t < 4; ++t)
          bf[t] = load_b_lds(&Bs[cur][((tb + t) * 16 + (lane & 15)) * KPAD +
                                      kh + ((lane >> 4) * 16)]);
#pragma unroll
        for (int t = 0; t < 4; ++t) acc[tb + t] = wmma_f16(a, bf[t], acc[tb + t]);
      }
    }

    if (more) {
      h8* lp = (h8*)&Bs[cur ^ 1][tid * KPAD];
#pragma unroll
      for (int i = 0; i < KSTEP / 8; ++i) lp[i] = pre[i];
      __syncthreads();
    }
  }

  const int hi  = lane >> 4;
  const int col = lane & 15;
#pragma unroll
  for (int t = 0; t < 8; ++t) {
#pragma unroll
    for (int r = 0; r < 8; ++r) {
      const int m = m0 + r + hi * 8;
      const int n = n0 + t * 16 + col;
      const float v = acc[t][r] + bias[n];
      if (mode == 2) {
        outf[(size_t)m * D_ + n] = v;
      } else {
        const int b = m >> 11, s = m & (S_ - 1);
        const int h = n >> 6,  hd = n & (HD_ - 1);
        if (mode == 0)
          outh[(((size_t)(b * H_ + h) * S_ + s) * HD_) + hd] = (_Float16)v;
        else
          outh[(((size_t)(b * H_ + h) * HD_ + hd) * S_) + s] = (_Float16)v;
      }
    }
  }
}

// ---------------------------------------------------------------------------
// Flash attention: 8-wave workgroup covers 128 q rows of one (b,h).
// Per 32-key step, K (32x64) and V^T (64x32) tiles are LDS-staged once and
// shared by all 8 waves (padded strides; batched frag loads).
__global__ __launch_bounds__(256)
void attn_wg(const _Float16* __restrict__ Q, const _Float16* __restrict__ K,
             const _Float16* __restrict__ Vt, _Float16* __restrict__ Y) {
  __shared__ __align__(16) _Float16 Ks[32 * KPAD];       // [key][kd], padded
  __shared__ __align__(16) _Float16 Vs[64 * VPAD];       // [dim][key], padded
  __shared__ __align__(16) _Float16 Ps[8][16 * VPAD];    // per-wave probs

  const int WGQ = S_ / 128;                              // 16 wg per (b,h)
  const int bh  = blockIdx.x / WGQ;
  const int qwg = (blockIdx.x % WGQ) * 128;

  const _Float16* Qb = Q  + (size_t)bh * S_ * HD_;
  const _Float16* Kb = K  + (size_t)bh * S_ * HD_;
  const _Float16* Vb = Vt + (size_t)bh * HD_ * S_;

  const int tid  = threadIdx.x;
  const int wave = tid >> 5;
  const int lane = tid & 31;
  const int hi   = lane >> 4;
  const int col  = lane & 15;
  const int qbase = qwg + wave * 16;

  const v16h aq0 = load_a_frag(Qb, HD_, qbase, 0,  lane);
  const v16h aq1 = load_a_frag(Qb, HD_, qbase, 32, lane);

  v8f acc[4];
#pragma unroll
  for (int t = 0; t < 4; ++t) acc[t] = (v8f){0.f,0.f,0.f,0.f,0.f,0.f,0.f,0.f};
  float mrow[8], lrow[8];
#pragma unroll
  for (int r = 0; r < 8; ++r) { mrow[r] = -1e30f; lrow[r] = 0.f; }

  const float scale = 0.125f;                            // 1/sqrt(64)
  const int steps = (qwg >> 5) + 4;                      // (qwg+128)/32 blocks

  for (int j = 0; j < steps; ++j) {
    const int jb = j * 32;

    // ---- cooperative stage: 256 threads x 32 B each
    {
      const int kr = tid >> 3, ko = (tid & 7) * 8;       // K: 32 rows x 64
      *(h8*)&Ks[kr * KPAD + ko] =
          *(const h8*)(Kb + (size_t)(jb + kr) * HD_ + ko);
      const int vr = tid >> 2, vo = (tid & 3) * 8;       // V^T: 64 rows x 32
      *(h8*)&Vs[vr * VPAD + vo] = *(const h8*)(Vb + (size_t)vr * S_ + jb + vo);
    }
    __syncthreads();

    if (jb <= qbase + 15) {                              // causal: wave has work
      // ---- scores from LDS K tile (4 frags batched, then 4 WMMAs)
      v8f s0 = (v8f){0.f,0.f,0.f,0.f,0.f,0.f,0.f,0.f};
      v8f s1 = s0;
      {
        v16h bk[4];
        bk[0] = load_b_lds(&Ks[(col)      * KPAD + hi * 16]);
        bk[1] = load_b_lds(&Ks[(col)      * KPAD + 32 + hi * 16]);
        bk[2] = load_b_lds(&Ks[(16 + col) * KPAD + hi * 16]);
        bk[3] = load_b_lds(&Ks[(16 + col) * KPAD + 32 + hi * 16]);
        s0 = wmma_f16(aq0, bk[0], s0);
        s0 = wmma_f16(aq1, bk[1], s0);
        s1 = wmma_f16(aq0, bk[2], s1);
        s1 = wmma_f16(aq1, bk[3], s1);
      }

      // ---- online softmax (rows split across 16-lane halves per C layout)
      _Float16* Pw = &Ps[wave][0];
#pragma unroll
      for (int r = 0; r < 8; ++r) {
        const int row = qbase + r + hi * 8;
        float v0 = (jb + col      <= row) ? s0[r] * scale : -1e30f;
        float v1 = (jb + 16 + col <= row) ? s1[r] * scale : -1e30f;
        float bm = fmaxf(v0, v1);
        bm = fmaxf(bm, __shfl_xor(bm, 1, 16));
        bm = fmaxf(bm, __shfl_xor(bm, 2, 16));
        bm = fmaxf(bm, __shfl_xor(bm, 4, 16));
        bm = fmaxf(bm, __shfl_xor(bm, 8, 16));
        const float mnew = fmaxf(mrow[r], bm);
        const float scl  = __expf(mrow[r] - mnew);
        const float p0   = __expf(v0 - mnew);
        const float p1   = __expf(v1 - mnew);
        float rs = p0 + p1;
        rs += __shfl_xor(rs, 1, 16);
        rs += __shfl_xor(rs, 2, 16);
        rs += __shfl_xor(rs, 4, 16);
        rs += __shfl_xor(rs, 8, 16);
        lrow[r] = lrow[r] * scl + rs;
        mrow[r] = mnew;
        acc[0][r] *= scl; acc[1][r] *= scl; acc[2][r] *= scl; acc[3][r] *= scl;
        Pw[(r + hi * 8) * VPAD + col]      = (_Float16)p0;
        Pw[(r + hi * 8) * VPAD + 16 + col] = (_Float16)p1;
      }
      // same-wave LDS store->load is in-order; fence compiler + DScnt
      asm volatile("s_wait_dscnt 0" ::: "memory");

      // ---- P (16x32) as A fragment from this wave's LDS slab
      const _Float16* pp = &Pw[col * VPAD + hi * 16];
      const v16h pa = cat8(*(const h8*)pp, *(const h8*)(pp + 8));

      // ---- acc += P * V from LDS V^T tile (4 frags batched, then 4 WMMAs)
      v16h bv[4];
#pragma unroll
      for (int t = 0; t < 4; ++t)
        bv[t] = load_b_lds(&Vs[(t * 16 + col) * VPAD + hi * 16]);
#pragma unroll
      for (int t = 0; t < 4; ++t) acc[t] = wmma_f16(pa, bv[t], acc[t]);
    }
    __syncthreads();                                     // protect Ks/Vs reuse
  }

  // ---- epilogue: y = acc / l, write f16 [B,S,D]
  const int b = bh >> 4, h = bh & 15;
#pragma unroll
  for (int r = 0; r < 8; ++r) {
    const float inv = 1.0f / lrow[r];
    const int row = qbase + r + hi * 8;
#pragma unroll
    for (int t = 0; t < 4; ++t) {
      const int d = h * HD_ + t * 16 + col;
      Y[((size_t)b * S_ + row) * D_ + d] = (_Float16)(acc[t][r] * inv);
    }
  }
}

// ---------------------------------------------------------------------------
extern "C" void kernel_launch(void* const* d_in, const int* in_sizes, int n_in,
                              void* d_out, int out_size, void* d_ws, size_t ws_size,
                              hipStream_t stream) {
  const float* x  = (const float*)d_in[0];
  const float* Wq = (const float*)d_in[1];
  const float* bq = (const float*)d_in[2];
  const float* Wk = (const float*)d_in[3];
  const float* bk = (const float*)d_in[4];
  const float* Wv = (const float*)d_in[5];
  const float* bv = (const float*)d_in[6];
  const float* Wp = (const float*)d_in[7];
  const float* bp = (const float*)d_in[8];
  float* out = (float*)d_out;

  // workspace carve-up (f16): X, 4 weights, Q, K, V^T, Y  -> 48 MB total
  _Float16* w = (_Float16*)d_ws;
  _Float16* Xh  = w;                 w += (size_t)MTOT * D_;
  _Float16* Wqh = w;                 w += (size_t)D_ * D_;
  _Float16* Wkh = w;                 w += (size_t)D_ * D_;
  _Float16* Wvh = w;                 w += (size_t)D_ * D_;
  _Float16* Wph = w;                 w += (size_t)D_ * D_;
  _Float16* Qh  = w;                 w += (size_t)MTOT * D_;
  _Float16* Kh  = w;                 w += (size_t)MTOT * D_;
  _Float16* Vth = w;                 w += (size_t)MTOT * D_;
  _Float16* Yh  = w;                 w += (size_t)MTOT * D_;
  (void)ws_size; (void)n_in; (void)in_sizes; (void)out_size;

  const int nX = MTOT * D_;
  const int nW = D_ * D_;
  cvt_f32_f16<<<(nX + 255) / 256, 256, 0, stream>>>(x,  Xh,  nX);
  cvt_f32_f16<<<(nW + 255) / 256, 256, 0, stream>>>(Wq, Wqh, nW);
  cvt_f32_f16<<<(nW + 255) / 256, 256, 0, stream>>>(Wk, Wkh, nW);
  cvt_f32_f16<<<(nW + 255) / 256, 256, 0, stream>>>(Wv, Wvh, nW);
  cvt_f32_f16<<<(nW + 255) / 256, 256, 0, stream>>>(Wp, Wph, nW);

  const int gemmBlocks = (MTOT / 64) * (D_ / 128);  // 512 wg x 128 thr
  gemm_wg<<<gemmBlocks, 128, 0, stream>>>(Xh, Wqh, bq, Qh,  nullptr, 0);
  gemm_wg<<<gemmBlocks, 128, 0, stream>>>(Xh, Wkh, bk, Kh,  nullptr, 0);
  gemm_wg<<<gemmBlocks, 128, 0, stream>>>(Xh, Wvh, bv, Vth, nullptr, 1);

  const int attnBlocks = B_ * H_ * (S_ / 128);      // 512 wg x 256 thr
  attn_wg<<<attnBlocks, 256, 0, stream>>>(Qh, Kh, Vth, Yh);

  gemm_wg<<<gemmBlocks, 128, 0, stream>>>(Yh, Wph, bp, nullptr, out, 2);
}
